// Mamba_39238821216919
// MI455X (gfx1250) — matrix operand order repaired and checked by reading, not compiled, not run on hardware
//
#include <hip/hip_runtime.h>
#include <hip/hip_bf16.h>

// ---------------------------------------------------------------------------
// Mamba forward for MI455X (gfx1250, wave32, WMMA + TDM).
//   D_MODEL=1024, D_STATE=16, D_CONV=4, DT_RANK=64, D_INNER=2048
//   B=2, L=2048 -> M = B*L = 4096 rows in every GEMM.
// GEMMs: v_wmma_f32_16x16x32_bf16, register-blocked 32x64 per wave
// (8 wmma / K-step). Weights + GEMM-only activations pre-converted to bf16
// -> hot loops do pure vector loads, no cvt. Scan: s[16] per lane in VGPRs;
// B/C chunks DMA'd to LDS by the Tensor Data Mover.
// ---------------------------------------------------------------------------

#define D_MODEL  1024
#define D_STATE  16
#define D_CONV   4
#define DT_RANK  64
#define D_INNER  2048
#define BATCH    2
#define SEQLEN   2048
#define M_ROWS   (BATCH * SEQLEN)     // 4096

typedef __attribute__((ext_vector_type(16))) __bf16        v16bf;
typedef __attribute__((ext_vector_type(8)))  __bf16        v8bf;
typedef __attribute__((ext_vector_type(8)))  float         v8f;
typedef __attribute__((ext_vector_type(4)))  unsigned int  u32x4;
typedef __attribute__((ext_vector_type(8)))  int           i32x8;
typedef __attribute__((ext_vector_type(4)))  int           i32x4;

// ---- fragment loaders (overloaded on source precision) --------------------
// A frag k order used here (contiguous k, not the ISA table order; B matches): lane<16 -> M=lane,    K = 0..15
//                      lane>=16-> M=lane-16, K = 16..31
// (Ap already points at  row, k0 + hi*16.)
__device__ __forceinline__ void load_a_frag(v16bf& a, const float* Ap) {
#pragma unroll
    for (int e = 0; e < 8; ++e) {
        a[e]     = (__bf16)Ap[e];
        a[e + 8] = (__bf16)Ap[e + 8];
    }
}
__device__ __forceinline__ void load_a_frag(v16bf& a, const __bf16* Ap) {
    const v8bf lo = *(const v8bf*)Ap;          // 16B aligned
    const v8bf hh = *(const v8bf*)(Ap + 8);
#pragma unroll
    for (int e = 0; e < 8; ++e) { a[e] = lo[e]; a[e + 8] = hh[e]; }
}
// B frag k order used here (contiguous k, same as A): N = lane%16, lane<16 -> K=0..15, lane>=16 -> K=16..31
// (Wp already points at  col, k0 + hi*16.)
__device__ __forceinline__ void load_b_frag(v16bf& b, const float* Wp) {
#pragma unroll
    for (int e = 0; e < 16; ++e) b[e] = (__bf16)Wp[e];
}
__device__ __forceinline__ void load_b_frag(v16bf& b, const __bf16* Wp) {
    b = *(const v16bf*)Wp;                     // 32B aligned
}

// ---------------------------------------------------------------------------
// C[M,N] = A[M,K] @ W[N,K]^T  (bf16 WMMA compute, f32 accumulate).
// blockDim = (32,4): 4 waves stacked along M; each wave owns a
// (MT*16) x (NT*16) strip -> MT*NT v_wmma per 32-wide K step.
// C/D 16x16 f32 layout: VGPR v -> M = v + 8*(lane>=16), N = lane%16.
// ---------------------------------------------------------------------------
template<typename TA, typename TW, int MT, int NT>
__global__ __launch_bounds__(128)
void gemm_wmma(const TA* __restrict__ A, int lda,
               const TW* __restrict__ W, int ldw,
               float* __restrict__ C, int ldc,
               int K)
{
    const int lane = threadIdx.x;                 // 0..31
    const int warp = threadIdx.y;                 // 0..3 (waves along M)
    const int hi   = lane >> 4;
    const int l15  = lane & 15;

    const int tile_m0 = (blockIdx.y * 4 + warp) * MT;
    const int tile_n0 = blockIdx.x * NT;

    v8f acc[MT][NT];
#pragma unroll
    for (int i = 0; i < MT; ++i)
#pragma unroll
        for (int j = 0; j < NT; ++j) acc[i][j] = (v8f){};

    for (int k0 = 0; k0 < K; k0 += 32) {
        v16bf afrag[MT];
#pragma unroll
        for (int i = 0; i < MT; ++i) {
            const TA* Ap = A + (long)((tile_m0 + i) * 16 + l15) * lda + k0 + hi * 16;
            load_a_frag(afrag[i], Ap);
            __builtin_prefetch(Ap + 32, 0, 1);    // speculative; OOB dropped
        }
        v16bf bfrag[NT];
#pragma unroll
        for (int j = 0; j < NT; ++j) {
            const TW* Wp = W + (long)((tile_n0 + j) * 16 + l15) * ldw + k0 + hi * 16;
            load_b_frag(bfrag[j], Wp);
            __builtin_prefetch(Wp + 32, 0, 1);
        }
#pragma unroll
        for (int i = 0; i < MT; ++i)
#pragma unroll
            for (int j = 0; j < NT; ++j)
                acc[i][j] = __builtin_amdgcn_wmma_f32_16x16x32_bf16(
                    false, afrag[i], false, bfrag[j],
                    (short)0, acc[i][j], false, false);
    }

#pragma unroll
    for (int i = 0; i < MT; ++i) {
        const int row0 = (tile_m0 + i) * 16 + hi * 8;
#pragma unroll
        for (int j = 0; j < NT; ++j) {
            const int col = (tile_n0 + j) * 16 + l15;
#pragma unroll
            for (int v = 0; v < 8; ++v)
                C[(long)(row0 + v) * ldc + col] = acc[i][j][v];
        }
    }
}

// ---------------------------------------------------------------------------
// f32 -> bf16 conversion (weights / GEMM-side activations), done once.
// ---------------------------------------------------------------------------
__global__ __launch_bounds__(256)
void cvt_bf16_kernel(const float* __restrict__ in, __bf16* __restrict__ out, int n)
{
    const int i = blockIdx.x * 256 + threadIdx.x;
    if (i < n) out[i] = (__bf16)in[i];
}

// ---------------------------------------------------------------------------
// Depthwise causal conv1d (taps=4) + SiLU.  u_raw = xz[..., :D_INNER].
// ---------------------------------------------------------------------------
__global__ __launch_bounds__(256)
void conv_silu_kernel(const float* __restrict__ xz,
                      const float* __restrict__ conv_w,
                      const float* __restrict__ conv_b,
                      float* __restrict__ u_act)
{
    int idx = blockIdx.x * 256 + threadIdx.x;          // over B*L*D_INNER
    if (idx >= BATCH * SEQLEN * D_INNER) return;
    const int d = idx % D_INNER;
    const int t = (idx / D_INNER) % SEQLEN;
    const int b = idx / (D_INNER * SEQLEN);

    float acc = conv_b[d];
#pragma unroll
    for (int j = 0; j < D_CONV; ++j) {
        const int tt = t - (D_CONV - 1) + j;
        if (tt >= 0)
            acc += conv_w[d * D_CONV + j] *
                   xz[((long)(b * SEQLEN + tt)) * (2 * D_INNER) + d];
    }
    u_act[((long)(b * SEQLEN + t)) * D_INNER + d] = acc / (1.0f + __expf(-acc));
}

// ---------------------------------------------------------------------------
// dt = softplus(dt_raw + b_dt)
// ---------------------------------------------------------------------------
__global__ __launch_bounds__(256)
void dt_softplus_kernel(float* __restrict__ dt_raw,
                        const float* __restrict__ b_dt)
{
    int idx = blockIdx.x * 256 + threadIdx.x;
    if (idx >= M_ROWS * D_INNER) return;
    const float x = dt_raw[idx] + b_dt[idx % D_INNER];
    dt_raw[idx] = (x > 20.0f) ? x : log1pf(__expf(x));
}

// ---------------------------------------------------------------------------
// Selective scan. One lane per (b, d) channel, s[16] in VGPRs.
// B/C for each 32-step chunk are a CONTIGUOUS 32-float slab of x_dbl
// (cols 64..95 of the 96-wide matrix) -> one 2D TDM tile per chunk:
//   tile_dim0 = 32 f32, tile_dim1 = 32 rows, tensor_dim0_stride = 96.
// Wave 0 issues tensor_load_to_lds (EXEC ignored for TDM), waits TENSORcnt,
// then all waves barrier. Emits y directly as bf16 for the final GEMM.
// grid = (D_INNER/256, BATCH), block = 256.
// ---------------------------------------------------------------------------
#define SCAN_CHUNK 32

__device__ __forceinline__
void tdm_load_bc_chunk(unsigned lds_off, const float* gsrc)
{
#if __has_builtin(__builtin_amdgcn_tensor_load_to_lds)
    const unsigned long long ga = (unsigned long long)gsrc;
    // D# group 0 (ISA 8.3): count=1 | lds_addr | global_addr[56:0] | type=2
    u32x4 g0;
    g0[0] = 1u;
    g0[1] = lds_off;
    g0[2] = (unsigned)(ga & 0xFFFFFFFFu);
    g0[3] = (unsigned)((ga >> 32) & 0x01FFFFFFu) | 0x80000000u;
    // D# group 1 (ISA 8.4): data_size=4B, dims/tile 32x32, stride0=96
    i32x8 g1;
    g1[0] = 0x00020000;            // workgroup_mask=0, data_size=2 (4B)
    g1[1] = (SCAN_CHUNK) << 16;    // tensor_dim0[15:0]  (bits 79:48)
    g1[2] = (SCAN_CHUNK) << 16;    // tensor_dim1[15:0]  (bits 111:80)
    g1[3] = (SCAN_CHUNK) << 16;    // tile_dim0          (bits 127:112)
    g1[4] = (SCAN_CHUNK);          // tile_dim1          (bits 143:128)
    g1[5] = 96;                    // tensor_dim0_stride low (bits 207:160)
    g1[6] = 0;
    g1[7] = 0;
#if __clang_major__ >= 23
    __builtin_amdgcn_tensor_load_to_lds(g0, g1, (i32x4)0, (i32x4)0, (i32x8)0, 0);
#else
    __builtin_amdgcn_tensor_load_to_lds(g0, g1, (i32x4)0, (i32x4)0, 0);
#endif
    __builtin_amdgcn_s_wait_tensorcnt(0);
#else
    (void)lds_off; (void)gsrc;
#endif
}

__global__ __launch_bounds__(256)
void scan_kernel(const float* __restrict__ dt,
                 const float* __restrict__ u,
                 const float* __restrict__ xdbl,   // [M_ROWS, 96]
                 const float* __restrict__ A_log,
                 const float* __restrict__ Dvec,
                 const float* __restrict__ xz,     // z = xz[..., D_INNER:]
                 __bf16* __restrict__ y)           // bf16: feeds final GEMM
{
    const int b = blockIdx.y;
    const int d = blockIdx.x * 256 + threadIdx.x;

    float Arow[D_STATE];
#pragma unroll
    for (int n = 0; n < D_STATE; ++n)
        Arow[n] = -__expf(A_log[d * D_STATE + n]);
    const float Dd = Dvec[d];

    float s[D_STATE];
#pragma unroll
    for (int n = 0; n < D_STATE; ++n) s[n] = 0.0f;

    // row t: cols 0..15 = B_ssm, cols 16..31 = C_ssm
    __shared__ float sBC[SCAN_CHUNK][2 * D_STATE];

    for (int t0 = 0; t0 < SEQLEN; t0 += SCAN_CHUNK) {
        __syncthreads();
#if __has_builtin(__builtin_amdgcn_tensor_load_to_lds)
        if (threadIdx.x < 32) {       // wave 0 only: single TDM issue
            tdm_load_bc_chunk((unsigned)(size_t)&sBC[0][0],
                              xdbl + (long)(b * SEQLEN + t0) * 96 + DT_RANK);
        }
#else
        for (int i = threadIdx.x; i < SCAN_CHUNK * 2 * D_STATE; i += 256) {
            const int tt = i / (2 * D_STATE), c = i % (2 * D_STATE);
            sBC[tt][c] = xdbl[(long)(b * SEQLEN + t0 + tt) * 96 + DT_RANK + c];
        }
#endif
        __syncthreads();

#pragma unroll 4
        for (int tt = 0; tt < SCAN_CHUNK; ++tt) {
            const int  t    = t0 + tt;
            const long base = (long)(b * SEQLEN + t) * D_INNER + d;
            const float dt_t = dt[base];
            const float u_t  = u[base];
            const float du   = dt_t * u_t;
            float acc = 0.0f;
#pragma unroll
            for (int n = 0; n < D_STATE; ++n) {
                const float dA = __expf(dt_t * Arow[n]);
                s[n] = s[n] * dA + du * sBC[tt][n];
                acc += s[n] * sBC[tt][D_STATE + n];
            }
            const float zv = xz[(long)(b * SEQLEN + t) * (2 * D_INNER) + D_INNER + d];
            const float sz = zv / (1.0f + __expf(-zv));
            y[base] = (__bf16)((acc + u_t * Dd) * sz);
        }
    }
}

// ---------------------------------------------------------------------------
extern "C" void kernel_launch(void* const* d_in, const int* in_sizes, int n_in,
                              void* d_out, int out_size, void* d_ws, size_t ws_size,
                              hipStream_t stream)
{
    const float* hidden = (const float*)d_in[0];  // [B, L, D_MODEL]
    const float* W_in   = (const float*)d_in[1];  // [2*D_INNER, D_MODEL]
    const float* conv_w = (const float*)d_in[2];  // [D_INNER, D_CONV]
    const float* conv_b = (const float*)d_in[3];  // [D_INNER]
    const float* W_x    = (const float*)d_in[4];  // [96, D_INNER]
    const float* W_dt   = (const float*)d_in[5];  // [D_INNER, DT_RANK]
    const float* b_dt   = (const float*)d_in[6];  // [D_INNER]
    const float* A_log  = (const float*)d_in[7];  // [D_INNER, D_STATE]
    const float* Dvec   = (const float*)d_in[8];  // [D_INNER]
    const float* W_out  = (const float*)d_in[9];  // [D_MODEL, D_INNER]
    float* out = (float*)d_out;                   // [B, L, D_MODEL]

    // ---- workspace layout --------------------------------------------------
    float*  xz     = (float*)d_ws;                            // 4096*4096 f32
    float*  u      = xz    + (size_t)M_ROWS * 2 * D_INNER;    // 4096*2048 f32
    float*  xdbl   = u     + (size_t)M_ROWS * D_INNER;        // 4096*96   f32
    float*  dtbuf  = xdbl  + (size_t)M_ROWS * 96;             // 4096*2048 f32
    __bf16* ybf    = (__bf16*)(dtbuf + (size_t)M_ROWS * D_INNER); // 4096*2048
    __bf16* hbf    = ybf    + (size_t)M_ROWS * D_INNER;       // 4096*1024
    __bf16* wbf_in = hbf    + (size_t)M_ROWS * D_MODEL;       // 4096*1024
    __bf16* wbf_x  = wbf_in + (size_t)2 * D_INNER * D_MODEL;  // 96*2048
    __bf16* wbf_dt = wbf_x  + (size_t)96 * D_INNER;           // 2048*64
    __bf16* wbf_out= wbf_dt + (size_t)D_INNER * DT_RANK;      // 1024*2048

    const dim3 gblk(32, 4);
    const int  gy = M_ROWS / 128;                 // 128 M-rows per block

    // 0) one-time f32 -> bf16 conversions (weights + hidden)
    cvt_bf16_kernel<<<(M_ROWS * D_MODEL + 255) / 256, 256, 0, stream>>>(
        hidden, hbf, M_ROWS * D_MODEL);
    cvt_bf16_kernel<<<(2 * D_INNER * D_MODEL + 255) / 256, 256, 0, stream>>>(
        W_in, wbf_in, 2 * D_INNER * D_MODEL);
    cvt_bf16_kernel<<<(96 * D_INNER + 255) / 256, 256, 0, stream>>>(
        W_x, wbf_x, 96 * D_INNER);
    cvt_bf16_kernel<<<(D_INNER * DT_RANK + 255) / 256, 256, 0, stream>>>(
        W_dt, wbf_dt, D_INNER * DT_RANK);
    cvt_bf16_kernel<<<(D_MODEL * D_INNER + 255) / 256, 256, 0, stream>>>(
        W_out, wbf_out, D_MODEL * D_INNER);

    // 1) xz = hidden @ W_in^T            (N=4096, K=1024, pure bf16 stream)
    gemm_wmma<__bf16, __bf16, 2, 4>
        <<<dim3((2 * D_INNER) / 64, gy), gblk, 0, stream>>>(
        hbf, D_MODEL, wbf_in, D_MODEL, xz, 2 * D_INNER, D_MODEL);

    // 2) u = silu(causal_conv1d(xz[..., :D_INNER]))
    conv_silu_kernel<<<(BATCH * SEQLEN * D_INNER + 255) / 256, 256, 0, stream>>>(
        xz, conv_w, conv_b, u);

    // 3) x_dbl = u @ W_x^T               (N=96, K=2048; f32 A, bf16 W)
    gemm_wmma<float, __bf16, 2, 2><<<dim3(96 / 32, gy), gblk, 0, stream>>>(
        u, D_INNER, wbf_x, D_INNER, xdbl, 96, D_INNER);

    // 4) dt_raw = dtr @ W_dt^T           (N=2048, K=64; dtr = xdbl[:, :64])
    gemm_wmma<float, __bf16, 2, 4><<<dim3(D_INNER / 64, gy), gblk, 0, stream>>>(
        xdbl, 96, wbf_dt, DT_RANK, dtbuf, D_INNER, DT_RANK);

    // 5) dt = softplus(dt_raw + b_dt)
    dt_softplus_kernel<<<(M_ROWS * D_INNER + 255) / 256, 256, 0, stream>>>(
        dtbuf, b_dt);

    // 6) selective scan + gating -> ybf   (TDM-staged B/C chunks, bf16 out)
    scan_kernel<<<dim3(D_INNER / 256, BATCH), 256, 0, stream>>>(
        dtbuf, u, xdbl, A_log, Dvec, xz, ybf);

    // 7) out = ybf @ W_out^T             (N=1024, K=2048, pure bf16 stream)
    gemm_wmma<__bf16, __bf16, 2, 4><<<dim3(D_MODEL / 64, gy), gblk, 0, stream>>>(
        ybf, D_INNER, wbf_out, D_INNER, out, D_MODEL, D_INNER);
}
